// Graph_encoder1_5815385719420
// MI455X (gfx1250) — matrix operand order repaired and checked by reading, not compiled, not run on hardware
//
#include <hip/hip_runtime.h>
#include <hip/hip_bf16.h>

// ---------------------------------------------------------------------------
// GAT-style graph encoder for MI455X (gfx1250, wave32, WMMA).
//
// Roofline: adj (80MB) sets the HBM floor (~3.5us at 23.3 TB/s); dominant
// compute is attention(10000x2000) @ kc_Wh(2000x256) = 10.2 GFLOP, done with
// v_wmma_f32_16x16x32_bf16 + hi/lo split (AhBh + AlBh + AhBl) for near-fp32
// accuracy at bf16 speed. kc_Wh and rd_w are pre-packed ONCE into the exact
// WMMA B-fragment LDS image (hi/lo bf16, K-pairs, 80B col stride) so the hot
// kernels stage B via raw 16B copies (global_load_async_to_lds_b128 when
// available -> ASYNCcnt overlap with the softmax v_exp_f32 work).
// ---------------------------------------------------------------------------

#define N_E   10000
#define N_KC  2000
#define ALPHA 0.2f
#define NEG_INF -9.0e15f

// Packed B chunk geometry: [hi|lo][256 cols][40 bf16 (32 valid K + 8 pad)].
// Col stride 80B: 16B-aligned for b128 copies and bank-conflict-free frag loads.
#define PK_STRIDE   40                         // bf16 per col
#define PK_CHUNK_BF (2 * 256 * PK_STRIDE)      // 20480 bf16 = 40KB per K-chunk
#define PK_CHUNK_U32 (PK_CHUNK_BF / 2)         // 10240 dwords

#if __has_builtin(__builtin_amdgcn_global_load_async_to_lds_b128)
#define HAVE_ASYNC_LDS 1
#else
#define HAVE_ASYNC_LDS 0
#endif

typedef __attribute__((ext_vector_type(16))) __bf16 v16bf;
typedef __attribute__((ext_vector_type(8)))  float  v8f;
typedef int v4i __attribute__((vector_size(16)));   // matches builtin param type
typedef __attribute__((address_space(1))) v4i* gptr_v4i;
typedef __attribute__((address_space(3))) v4i* lptr_v4i;

__device__ __forceinline__ float lrelu(float x) { return x > 0.f ? x : ALPHA * x; }

__device__ __forceinline__ v8f wmma_bf16(v16bf a, v16bf b, v8f c) {
  return __builtin_amdgcn_wmma_f32_16x16x32_bf16(false, a, false, b, (short)0, c,
                                                 false, false);
}

// 16B global->LDS copy; async (ASYNCcnt) on CDNA5 when the builtin exists.
__device__ __forceinline__ void copy16B_to_lds(const void* g, void* l) {
#if HAVE_ASYNC_LDS
  __builtin_amdgcn_global_load_async_to_lds_b128(
      (gptr_v4i)(unsigned long long)(size_t)g,
      (lptr_v4i)(unsigned)(size_t)l, 0, 0);
#else
  *(uint4*)l = *(const uint4*)g;
#endif
}
__device__ __forceinline__ void lds_copy_fence() {
#if HAVE_ASYNC_LDS
  asm volatile("s_wait_asynccnt 0" ::: "memory");
#endif
}

// A fragment (16x32 bf16, hi/lo split) from a [16][33] f32 LDS tile.
// Lane (m=lane&15, half=lane>>4); pair e holds K=half*16+2e.
__device__ __forceinline__ void build_a_frag(const float As[16][33], int m, int half,
                                             v16bf& ah, v16bf& al) {
#pragma unroll
  for (int e = 0; e < 8; ++e) {
    int kb = half * 16 + 2 * e;
    float x0 = As[m][kb], x1 = As[m][kb + 1];
    __bf16 h0 = (__bf16)x0, h1 = (__bf16)x1;
    ah[2 * e]     = h0;
    ah[2 * e + 1] = h1;
    al[2 * e]     = (__bf16)(x0 - (float)h0);
    al[2 * e + 1] = (__bf16)(x1 - (float)h1);
  }
}

// B fragment (32x16 bf16) from packed LDS image Bx[col][kk].
// Lane (n=col, half); pair e holds K = half*16 + 2e.
__device__ __forceinline__ void build_b_frag(const __bf16 (*Bx)[PK_STRIDE], int col,
                                             int half, v16bf& b) {
#pragma unroll
  for (int e = 0; e < 8; ++e) {
    int kk = half * 16 + 2 * e;
    b[2 * e]     = Bx[col][kk];
    b[2 * e + 1] = Bx[col][kk + 1];
  }
}

// Pack an (even,odd) K-pair of f32 into hi/lo bf16 dwords.
__device__ __forceinline__ void pack_pair(float x0, float x1, unsigned& hi,
                                          unsigned& lo) {
  __bf16 h0 = (__bf16)x0, h1 = (__bf16)x1;
  __bf16 l0 = (__bf16)(x0 - (float)h0), l1 = (__bf16)(x1 - (float)h1);
  hi = (unsigned)__builtin_bit_cast(unsigned short, h0) |
       ((unsigned)__builtin_bit_cast(unsigned short, h1) << 16);
  lo = (unsigned)__builtin_bit_cast(unsigned short, l0) |
       ((unsigned)__builtin_bit_cast(unsigned short, l1) << 16);
}

// ---------------------------------------------------------------------------
// Kernel 1: w1a1 = W1 @ a[:256], w1a2 = W1 @ a[256:]
// ---------------------------------------------------------------------------
__global__ __launch_bounds__(256) void prep_kernel(const float* __restrict__ W1,
                                                   const float* __restrict__ a,
                                                   float* __restrict__ w1a1,
                                                   float* __restrict__ w1a2) {
  int k = threadIdx.x;
  float h1 = 0.f, h2 = 0.f;
  for (int o = 0; o < 256; ++o) {
    float w = W1[k * 256 + o];
    h1 += w * a[o];
    h2 += w * a[256 + o];
  }
  w1a1[k] = h1;
  w1a2[k] = h2;
}

// ---------------------------------------------------------------------------
// Kernel 2: s_ex[i] = ex[i,:]·w1a1 ; s_kc[j] = kc[j,:]·w1a2  (wave per row)
// ---------------------------------------------------------------------------
__global__ __launch_bounds__(256) void score_kernel(const float* __restrict__ ex,
                                                    const float* __restrict__ kc,
                                                    const float* __restrict__ w1a1,
                                                    const float* __restrict__ w1a2,
                                                    float* __restrict__ sex,
                                                    float* __restrict__ skc) {
  int lane = threadIdx.x & 31, wave = threadIdx.x >> 5;
  int row = blockIdx.x * 8 + wave;
  if (row >= N_E + N_KC) return;
  const float* src;
  const float* vec;
  if (row < N_E) { src = ex + (size_t)row * 256;          vec = w1a1; }
  else           { src = kc + (size_t)(row - N_E) * 256;  vec = w1a2; }
  float s = 0.f;
#pragma unroll
  for (int u = 0; u < 8; ++u) {
    int c = lane + u * 32;
    s += src[c] * vec[c];
  }
  for (int off = 16; off > 0; off >>= 1) s += __shfl_xor(s, off, 32);
  if (lane == 0) {
    if (row < N_E) sex[row] = s;
    else           skc[row - N_E] = s;
  }
}

// ---------------------------------------------------------------------------
// Kernel 3: zero-fill helper (workspace is poisoned, K-tail must be 0).
// ---------------------------------------------------------------------------
__global__ __launch_bounds__(256) void zero_kernel(unsigned* __restrict__ p, int n) {
  int i = blockIdx.x * 256 + threadIdx.x;
  if (i < n) p[i] = 0u;
}

// ---------------------------------------------------------------------------
// Kernel 4: pack rd_w (256x512 row-major) -> B-fragment image of rd_w^T,
// 16 K-chunks of 32, hi/lo split. One block per chunk.
// ---------------------------------------------------------------------------
__global__ __launch_bounds__(256) void pack_rdw_kernel(const float* __restrict__ rd_w,
                                                       unsigned* __restrict__ pk) {
  int ck = blockIdx.x, n = threadIdx.x;
  unsigned* hib = pk + (size_t)ck * PK_CHUNK_U32 + n * (PK_STRIDE / 2);
  unsigned* lob = hib + 256 * (PK_STRIDE / 2);
#pragma unroll
  for (int vk = 0; vk < 16; ++vk) {
    int k = ck * 32 + 2 * vk;
    unsigned hi, lo;
    pack_pair(rd_w[n * 512 + k], rd_w[n * 512 + k + 1], hi, lo);
    hib[vk] = hi;
    lob[vk] = lo;
  }
}

// ---------------------------------------------------------------------------
// Kernel 5: C(Mx256) = A(Mx256) @ B(256x256) via split-bf16 WMMA.
// Epilogue: either f32 C, or directly the packed hi/lo B-fragment image
// (used for kc_Wh, whose only consumer is the attention GEMM).
// ---------------------------------------------------------------------------
__global__ __launch_bounds__(256) void gemm256_kernel(const float* __restrict__ A,
                                                      const float* __restrict__ B,
                                                      float* __restrict__ C,
                                                      unsigned* __restrict__ Cpk,
                                                      int M) {
  __shared__ float  As[16][33];
  __shared__ __bf16 Bhi[256][34];
  __shared__ __bf16 Blo[256][34];
  const int t = threadIdx.x;
  const int lane = t & 31, wave = t >> 5;
  const int m = lane & 15, half = lane >> 4;
  const int row0 = blockIdx.x * 16;
  v8f acc0 = {}, acc1 = {};

  for (int k0 = 0; k0 < 256; k0 += 32) {
    __syncthreads();
#pragma unroll
    for (int q = 0; q < 2; ++q) {  // stage A 16x32
      int idx = q * 256 + t;
      int r = idx >> 5, c = idx & 31;
      int gr = row0 + r;
      As[r][c] = (gr < M) ? A[(size_t)gr * 256 + k0 + c] : 0.f;
    }
#pragma unroll 8
    for (int r = 0; r < 32; ++r) {  // stage B 32x256 transposed + split
      float val = B[(k0 + r) * 256 + t];
      __bf16 h = (__bf16)val;
      Bhi[t][r] = h;
      Blo[t][r] = (__bf16)(val - (float)h);
    }
    __syncthreads();
    v16bf ah, al;
    build_a_frag(As, m, half, ah, al);
#pragma unroll
    for (int nt = 0; nt < 2; ++nt) {
      int col = (wave * 2 + nt) * 16 + m;
      v16bf bh, bl;
#pragma unroll
      for (int e = 0; e < 8; ++e) {
        int kk = half * 16 + 2 * e;
        bh[2 * e] = Bhi[col][kk];
        bh[2 * e + 1] = Bhi[col][kk + 1];
        bl[2 * e] = Blo[col][kk];
        bl[2 * e + 1] = Blo[col][kk + 1];
      }
      v8f& acc = nt ? acc1 : acc0;
      acc = wmma_bf16(ah, bh, acc);
      acc = wmma_bf16(al, bh, acc);
      acc = wmma_bf16(ah, bl, acc);
    }
  }

  if (Cpk) {  // packed epilogue: rows of C are the K dim of the next GEMM
    int jb = row0 & 31, chunk = row0 >> 5;
    unsigned* hib = Cpk + (size_t)chunk * PK_CHUNK_U32;
    unsigned* lob = hib + 256 * (PK_STRIDE / 2);
#pragma unroll
    for (int nt = 0; nt < 2; ++nt) {
      int col = (wave * 2 + nt) * 16 + m;
      v8f acc = nt ? acc1 : acc0;
#pragma unroll
      for (int u = 0; u < 4; ++u) {
        int gr = row0 + 2 * u + 8 * half;  // even row of the pair
        if (gr < M) {
          unsigned hi, lo;
          pack_pair(acc[2 * u], acc[2 * u + 1], hi, lo);
          int vk = (jb + 2 * u + 8 * half) >> 1;
          hib[col * (PK_STRIDE / 2) + vk] = hi;
          lob[col * (PK_STRIDE / 2) + vk] = lo;
        }
      }
    }
  } else {
#pragma unroll
    for (int nt = 0; nt < 2; ++nt) {
      int col = (wave * 2 + nt) * 16 + m;
      v8f acc = nt ? acc1 : acc0;
#pragma unroll
      for (int e = 0; e < 8; ++e) {
        int gr = row0 + e + 8 * half;  // C layout: M = v + 8*(lane/16)
        if (gr < M) C[(size_t)gr * 256 + col] = acc[e];
      }
    }
  }
}

// ---------------------------------------------------------------------------
// Kernel 6: fused masked softmax + attention @ kc_Wh.
// Block = 16 exercise rows. Phase 1: online masked softmax stats (adj pass 1).
// Phase 2 per 32-K chunk: async-copy pre-packed kc_Wh image into LDS while
// computing the probability tile (adj pass 2 hits L2), then 6 WMMAs/wave.
// ---------------------------------------------------------------------------
__global__ __launch_bounds__(256) void attn_kernel(const int* __restrict__ adj,
                                                   const float* __restrict__ s_ex,
                                                   const float* __restrict__ s_kc,
                                                   const unsigned* __restrict__ kcpk,
                                                   float* __restrict__ newkc) {
  __shared__ float  As[16][33];
  __shared__ alignas(16) __bf16 Bbuf[2][256][PK_STRIDE];  // 40KB packed image
  __shared__ float  skc_l[N_KC];
  __shared__ float  s_mx[16], s_inv[16], s_se[16];
  const int t = threadIdx.x;
  const int lane = t & 31, wave = t >> 5;
  const int m = lane & 15, half = lane >> 4;
  const int row0 = blockIdx.x * 16;

  for (int j = t; j < N_KC; j += 256) skc_l[j] = s_kc[j];
  if (t < 16) s_se[t] = s_ex[row0 + t];
  __syncthreads();

  // Phase 1: online softmax stats; wave w owns rows 2w, 2w+1.
  for (int rr = 0; rr < 2; ++rr) {
    int r = wave * 2 + rr;
    int gi = row0 + r;
    float se = s_se[r];
    float mx = -3.0e38f, sm = 0.f;
    for (int j = lane; j < N_KC; j += 32) {
      float ev = (adj[(size_t)gi * N_KC + j] > 0) ? lrelu(se + skc_l[j]) : NEG_INF;
      float nm = fmaxf(mx, ev);
      sm = sm * __expf(mx - nm) + __expf(ev - nm);
      mx = nm;
    }
    for (int off = 16; off > 0; off >>= 1) {
      float mo = __shfl_xor(mx, off, 32);
      float so = __shfl_xor(sm, off, 32);
      float nm = fmaxf(mx, mo);
      sm = sm * __expf(mx - nm) + so * __expf(mo - nm);
      mx = nm;
    }
    if (lane == 0) { s_mx[r] = mx; s_inv[r] = 1.f / sm; }
  }
  __syncthreads();

  v8f acc0 = {}, acc1 = {};
  for (int ck = 0; ck < 63; ++ck) {  // 63 chunks of 32 cover 2000 (tail pre-zeroed)
    int k0 = ck * 32;
    __syncthreads();
    // Issue async copy of the packed B chunk first so it overlaps the exp math.
    {
      const unsigned* gsrc = kcpk + (size_t)ck * PK_CHUNK_U32;
      unsigned* ldst = (unsigned*)&Bbuf[0][0][0];
      for (int i = t; i < PK_CHUNK_U32 / 4; i += 256)
        copy16B_to_lds(gsrc + i * 4, ldst + i * 4);
    }
#pragma unroll
    for (int q = 0; q < 2; ++q) {  // probability tile 16x32 (coalesced adj reads)
      int idx = q * 256 + t;
      int r = idx >> 5, c = idx & 31;
      int j = k0 + c;
      float p = 0.f;
      if (j < N_KC) {
        int gi = row0 + r;
        float ev = (adj[(size_t)gi * N_KC + j] > 0) ? lrelu(s_se[r] + skc_l[j])
                                                    : NEG_INF;
        p = __expf(ev - s_mx[r]) * s_inv[r];  // all-masked row -> uniform 1/2000
      }
      As[r][c] = p;
    }
    lds_copy_fence();
    __syncthreads();
    v16bf ah, al;
    build_a_frag(As, m, half, ah, al);
#pragma unroll
    for (int nt = 0; nt < 2; ++nt) {
      int col = (wave * 2 + nt) * 16 + m;
      v16bf bh, bl;
      build_b_frag(Bbuf[0], col, half, bh);
      build_b_frag(Bbuf[1], col, half, bl);
      v8f& acc = nt ? acc1 : acc0;
      acc = wmma_bf16(ah, bh, acc);
      acc = wmma_bf16(al, bh, acc);
      acc = wmma_bf16(ah, bl, acc);
    }
  }
#pragma unroll
  for (int nt = 0; nt < 2; ++nt) {
    int col = (wave * 2 + nt) * 16 + m;
    v8f acc = nt ? acc1 : acc0;
#pragma unroll
    for (int e = 0; e < 8; ++e) {
      int gr = row0 + e + 8 * half;
      newkc[(size_t)gr * 256 + col] = acc[e];
    }
  }
}

// ---------------------------------------------------------------------------
// Kernel 7: out = elu(feat @ rd_w^T + rd_b), feat = [new_kc | new_kc*ex_Eh]
// built on the fly; B comes from the pre-packed rd_w image (async copy).
// ---------------------------------------------------------------------------
__global__ __launch_bounds__(256) void final_kernel(const float* __restrict__ nk,
                                                    const float* __restrict__ exE,
                                                    const unsigned* __restrict__ rdwpk,
                                                    const float* __restrict__ rd_b,
                                                    float* __restrict__ out) {
  __shared__ float  As[16][33];
  __shared__ alignas(16) __bf16 Bbuf[2][256][PK_STRIDE];
  const int t = threadIdx.x;
  const int lane = t & 31, wave = t >> 5;
  const int m = lane & 15, half = lane >> 4;
  const int row0 = blockIdx.x * 16;
  v8f acc0 = {}, acc1 = {};

  for (int ck = 0; ck < 16; ++ck) {  // K = 512
    int k0 = ck * 32;
    __syncthreads();
    {
      const unsigned* gsrc = rdwpk + (size_t)ck * PK_CHUNK_U32;
      unsigned* ldst = (unsigned*)&Bbuf[0][0][0];
      for (int i = t; i < PK_CHUNK_U32 / 4; i += 256)
        copy16B_to_lds(gsrc + i * 4, ldst + i * 4);
    }
#pragma unroll
    for (int q = 0; q < 2; ++q) {  // stage feat tile 16x32
      int idx = q * 256 + t;
      int r = idx >> 5, c = idx & 31;
      int k = k0 + c, gr = row0 + r;
      float v;
      if (k < 256) v = nk[(size_t)gr * 256 + k];
      else {
        int kk = k - 256;
        v = nk[(size_t)gr * 256 + kk] * exE[(size_t)gr * 256 + kk];
      }
      As[r][c] = v;
    }
    lds_copy_fence();
    __syncthreads();
    v16bf ah, al;
    build_a_frag(As, m, half, ah, al);
#pragma unroll
    for (int nt = 0; nt < 2; ++nt) {
      int col = (wave * 2 + nt) * 16 + m;
      v16bf bh, bl;
      build_b_frag(Bbuf[0], col, half, bh);
      build_b_frag(Bbuf[1], col, half, bl);
      v8f& acc = nt ? acc1 : acc0;
      acc = wmma_bf16(ah, bh, acc);
      acc = wmma_bf16(al, bh, acc);
      acc = wmma_bf16(ah, bl, acc);
    }
  }
#pragma unroll
  for (int nt = 0; nt < 2; ++nt) {
    int col = (wave * 2 + nt) * 16 + m;
    v8f acc = nt ? acc1 : acc0;
    float b = rd_b[col];
#pragma unroll
    for (int e = 0; e < 8; ++e) {
      int gr = row0 + e + 8 * half;
      float val = acc[e] + b;
      out[(size_t)gr * 256 + col] = val > 0.f ? val : (__expf(val) - 1.f);
    }
  }
}

// ---------------------------------------------------------------------------
extern "C" void kernel_launch(void* const* d_in, const int* in_sizes, int n_in,
                              void* d_out, int out_size, void* d_ws, size_t ws_size,
                              hipStream_t stream) {
  (void)in_sizes; (void)n_in; (void)out_size; (void)ws_size;
  const float* ex   = (const float*)d_in[0];  // (10000,256)
  const float* kc   = (const float*)d_in[1];  // (2000,256)
  const int*   adj  = (const int*)d_in[2];    // (10000,2000)
  const float* W1   = (const float*)d_in[3];  // (256,256)
  const float* E    = (const float*)d_in[4];  // (256,256)
  const float* a    = (const float*)d_in[5];  // (512,1)
  const float* rd_w = (const float*)d_in[6];  // (256,512)
  const float* rd_b = (const float*)d_in[7];  // (256,)

  float* ws      = (float*)d_ws;
  float* w1a1    = ws;                              // 256
  float* w1a2    = w1a1 + 256;                      // 256
  float* sex     = w1a2 + 256;                      // 10000
  float* skc     = sex + N_E;                       // 2000
  unsigned* kcpk = (unsigned*)(skc + N_KC);         // 63 * 10240 dwords
  unsigned* rdwpk = kcpk + (size_t)63 * PK_CHUNK_U32;  // 16 * 10240 dwords
  float* exEh    = (float*)(rdwpk + (size_t)16 * PK_CHUNK_U32);  // 10000*256
  float* newkc   = exEh + (size_t)N_E * 256;        // 10000*256
  // total ~23.8 MB of workspace

  prep_kernel   <<<1, 256, 0, stream>>>(W1, a, w1a1, w1a2);
  score_kernel  <<<(N_E + N_KC) / 8, 256, 0, stream>>>(ex, kc, w1a1, w1a2, sex, skc);
  // zero the K-tail chunk (j = 2000..2015) of the packed kc_Wh image
  zero_kernel   <<<(PK_CHUNK_U32 + 255) / 256, 256, 0, stream>>>(
      kcpk + (size_t)62 * PK_CHUNK_U32, PK_CHUNK_U32);
  pack_rdw_kernel<<<16, 256, 0, stream>>>(rd_w, rdwpk);
  gemm256_kernel<<<N_KC / 16, 256, 0, stream>>>(kc, W1, nullptr, kcpk, N_KC);
  gemm256_kernel<<<N_E / 16, 256, 0, stream>>>(ex, E, exEh, nullptr, N_E);
  attn_kernel   <<<N_E / 16, 256, 0, stream>>>(adj, sex, skc, kcpk, newkc);
  final_kernel  <<<N_E / 16, 256, 0, stream>>>(newkc, exEh, rdwpk, rd_b, (float*)d_out);
}
